// InterContrastiveLoss_14491219657438
// MI455X (gfx1250) — matrix operand order repaired and checked by reading, not compiled, not run on hardware
//
#include <hip/hip_runtime.h>
#include <hip/hip_bf16.h>

// ---------------- static problem config (matches reference setup) -----------
#define S_DIM 64
#define C_DIM 256
#define N_DIM 64
#define B_DIM 16
#define M_DIM 128
#define P_DIM 2080          // triu(64x64)
#define NN    4096          // N*N
#define PT    130           // P/16 tiles
#define GRP   17            // ceil(130/8) p-tile groups per video
#define NBLK  (B_DIM*GRP)   // 272 blocks for big GEMM
#define INV_T 10.0f         // 1/temperature
#define NEG_IOU 0.5f
#define SF_LDS_STRIDE 264   // 256 halves + 8 halves (16B) pad -> bank-conflict-free

typedef __attribute__((ext_vector_type(16))) _Float16 v16h;
typedef __attribute__((ext_vector_type(8)))  _Float16 v8h;
typedef __attribute__((ext_vector_type(8)))  float    v8f;
typedef __attribute__((ext_vector_type(4)))  unsigned int u32x4;
typedef __attribute__((ext_vector_type(8)))  int      i32x8;
typedef __attribute__((ext_vector_type(4)))  int      i32x4;

// ---------------- workspace layout (bytes, all 256B aligned) ----------------
#define OFF_FLATRC   0u          // int[2080]
#define OFF_B2S      8448u       // int[16]
#define OFF_SCNT     8704u       // int[16]
#define OFF_M2S      8960u       // int[128]
#define OFF_TOPK     10240u      // int[128]
#define OFF_DMAT     11008u      // float[128*64]
#define OFF_PARTIAL  44032u      // float[272*64]
#define OFF_NEGQ     113664u     // float[64]
#define OFF_SFH      114176u     // half[64*256]
#define OFF_TVH      146944u     // half[128*256]
#define OFF_VFH      212480u     // half[16*2080*256]  (~17 MB)

// ---------------- K0: metadata ----------------------------------------------
__global__ void init_meta(const int* __restrict__ num_sent,
                          const int* __restrict__ num_tgt,
                          int* __restrict__ flat_rc,
                          int* __restrict__ b2s,
                          int* __restrict__ scnt,
                          int* __restrict__ m2s) {
  if (threadIdx.x != 0 || blockIdx.x != 0) return;
  int p = 0;
  for (int r = 0; r < N_DIM; ++r)
    for (int c = r; c < N_DIM; ++c) flat_rc[p++] = r * N_DIM + c;
  int acc = 0;
  for (int b = 0; b < B_DIM; ++b) {
    b2s[b]  = acc;
    scnt[b] = num_sent[b];
    acc += num_sent[b];
  }
  int mi = 0;
  for (int s = 0; s < S_DIM; ++s)
    for (int k = 0; k < num_tgt[s]; ++k) m2s[mi++] = s;
}

// ---------------- block reduction helper ------------------------------------
__device__ __forceinline__ float block_sum256(float v, float* red) {
  int t = threadIdx.x;
  red[t] = v;
  __syncthreads();
  for (int off = 128; off > 0; off >>= 1) {
    if (t < off) red[t] += red[t + off];
    __syncthreads();
  }
  return red[0];
}

// ---------------- K1: normalize sentence feats -> f16 -----------------------
__global__ __launch_bounds__(256) void norm_sents(const float* __restrict__ sf,
                                                  _Float16* __restrict__ sfh) {
  __shared__ float red[256];
  int s = blockIdx.x, c = threadIdx.x;
  float x  = sf[s * C_DIM + c];
  float ss = block_sum256(x * x, red);
  float n  = fmaxf(sqrtf(ss), 1e-12f);
  sfh[s * C_DIM + c] = (_Float16)(x / n);
}

// ---------------- K2: gather + normalize selected videos -> packed f16 A ----
__global__ __launch_bounds__(256) void norm_video(const float* __restrict__ vfeats,
                                                  const int* __restrict__ b2s,
                                                  const int* __restrict__ flat_rc,
                                                  _Float16* __restrict__ vfh) {
  __shared__ float red[256];
  int bb = blockIdx.x / P_DIM;
  int p  = blockIdx.x % P_DIM;
  int s  = b2s[bb];
  int fl = flat_rc[p];
  int c  = threadIdx.x;
  float x  = vfeats[((size_t)s * C_DIM + c) * NN + fl];
  float ss = block_sum256(x * x, red);
  float n  = fmaxf(sqrtf(ss), 1e-12f);
  vfh[((size_t)bb * P_DIM + p) * C_DIM + c] = (_Float16)(x / n);
}

// ---------------- K3: argmax of iou2ds over masked proposals ----------------
__global__ __launch_bounds__(256) void argmax_iou(const float* __restrict__ iou2ds,
                                                  const int* __restrict__ flat_rc,
                                                  int* __restrict__ topk_flat) {
  __shared__ float bv[256];
  __shared__ int   bi[256];
  int m = blockIdx.x, t = threadIdx.x;
  float best = -1e30f;
  int   bidx = 0x3fffffff;
  for (int p = t; p < P_DIM; p += 256) {
    float v = iou2ds[(size_t)m * NN + flat_rc[p]];
    if (v > best) { best = v; bidx = p; }
  }
  bv[t] = best; bi[t] = bidx;
  __syncthreads();
  for (int off = 128; off > 0; off >>= 1) {
    if (t < off) {
      float v2 = bv[t + off]; int i2 = bi[t + off];
      if (v2 > bv[t] || (v2 == bv[t] && i2 < bi[t])) { bv[t] = v2; bi[t] = i2; }
    }
    __syncthreads();
  }
  if (t == 0) topk_flat[m] = flat_rc[bi[0]];
}

// ---------------- K4: gather + normalize top-k features -> f16 --------------
__global__ __launch_bounds__(256) void norm_topk(const float* __restrict__ vfeats,
                                                 const int* __restrict__ m2s,
                                                 const int* __restrict__ topk_flat,
                                                 _Float16* __restrict__ tvh) {
  __shared__ float red[256];
  int m  = blockIdx.x;
  int s  = m2s[m];
  int fl = topk_flat[m];
  int c  = threadIdx.x;
  float x  = vfeats[((size_t)s * C_DIM + c) * NN + fl];
  float ss = block_sum256(x * x, red);
  float n  = fmaxf(sqrtf(ss), 1e-12f);
  tvh[m * C_DIM + c] = (_Float16)(x / n);
}

// ---------------- WMMA fragment loaders -------------------------------------
// A (16x32 f16): lanes 0-15 row M=lane, K = koff+0..15;
//                lanes 16-31 row M=lane-16, K = koff+16..31
__device__ __forceinline__ v16h load_a_frag(const _Float16* __restrict__ arow,
                                            int hi, int koff) {
  int ka = koff + hi * 16;
  v8h lo = *(const v8h*)(arow + ka);
  v8h hh = *(const v8h*)(arow + ka + 8);
  return __builtin_shufflevector(lo, hh, 0,1,2,3,4,5,6,7,8,9,10,11,12,13,14,15);
}
// B (32x16 f16): lane column N=lane&15; lanes 0-15 K=koff+0..15, lanes16-31 +16
__device__ __forceinline__ v16h load_b_frag(const _Float16* __restrict__ sfh,
                                            int col, int hi, int koff) {
  return *(const v16h*)(sfh + col * C_DIM + koff + hi * 16);
}
// Same, from LDS-staged copy (row stride SF_LDS_STRIDE, 16B-safe loads)
__device__ __forceinline__ v16h load_b_lds(const _Float16* sfs,
                                           int col, int hi, int koff) {
  const _Float16* p = sfs + col * SF_LDS_STRIDE + koff + hi * 16;
  v8h lo = *(const v8h*)(p);
  v8h hh = *(const v8h*)(p + 8);
  return __builtin_shufflevector(lo, hh, 0,1,2,3,4,5,6,7,8,9,10,11,12,13,14,15);
}

// ---------------- TDM staging of SF (64x256 f16 = 32KB) into LDS ------------
// D#: 1D tile of 8192 dwords; pad_enable inserts 4 dwords (16B) after every
// 128 dwords (512B = one SF row) -> LDS row stride 528B (SF_LDS_STRIDE halves).
__device__ __forceinline__ void stage_sf_lds(const _Float16* __restrict__ sfh,
                                             _Float16* sfs) {
#if __has_builtin(__builtin_amdgcn_tensor_load_to_lds)
  if ((threadIdx.x >> 5) == 0) {          // one wave issues the TDM op
    unsigned int lds_off = (unsigned int)(uintptr_t)sfs;  // low 32 = LDS addr
    unsigned long long ga = (unsigned long long)(uintptr_t)sfh;
    u32x4 g0;
    g0[0] = 1u;                                           // count=1 valid D#
    g0[1] = lds_off;                                      // lds_addr
    g0[2] = (unsigned int)ga;                             // global_addr[31:0]
    g0[3] = (unsigned int)((ga >> 32) & 0x01ffffffu)      // global_addr[56:32]
            | 0x80000000u;                                // type=2 (image)
    i32x8 g1;
    g1[0] = (int)((2u << 16)      // data_size = 4B
                | (1u << 20)      // pad_enable
                | (6u << 22)      // pad_interval: 128 dwords (512B)
                | (3u << 25));    // pad_amount: 4 dwords (16B)
    g1[1] = (int)(8192u << 16);   // tensor_dim0 = 8192 dwords (low 16)
    g1[2] = (int)(1u << 16);      // tensor_dim0 hi | tensor_dim1 = 1
    g1[3] = (int)(8192u << 16);   // tensor_dim1 hi | tile_dim0 = 8192
    g1[4] = 1;                    // tile_dim1 = 1, tile_dim2 = 0
    g1[5] = 8192;                 // tensor_dim0_stride[31:0]
    g1[6] = 0;
    g1[7] = 0;
    i32x4 gz = {0, 0, 0, 0};
#if __clang_major__ >= 23
    i32x8 gz8 = {0, 0, 0, 0, 0, 0, 0, 0};
    __builtin_amdgcn_tensor_load_to_lds(g0, g1, gz, gz, gz8, 0);
#else
    __builtin_amdgcn_tensor_load_to_lds(g0, g1, gz, gz, 0);
#endif
    __builtin_amdgcn_s_wait_tensorcnt(0);
  }
#else
  for (int i = threadIdx.x; i < S_DIM * C_DIM; i += 256) {
    int r = i >> 8, c = i & 255;
    sfs[r * SF_LDS_STRIDE + c] = sfh[i];
  }
#endif
  __syncthreads();
}

// ---------------- K5: small WMMA GEMM  D(MxS) = TV * SF^T -------------------
__global__ __launch_bounds__(256) void gemm_topk(const _Float16* __restrict__ tvh,
                                                 const _Float16* __restrict__ sfh,
                                                 float* __restrict__ D) {
  int wave = threadIdx.x >> 5;
  int lane = threadIdx.x & 31;
  int hi   = lane >> 4;
  int ln   = lane & 15;
  int mtile = wave;                       // 8 waves -> 8 m-tiles of 16 = 128
  const _Float16* arow = tvh + (size_t)(mtile * 16 + ln) * C_DIM;
  v8f acc[4] = {};
  for (int kk = 0; kk < C_DIM / 32; ++kk) {
    int koff = kk * 32;
    v16h a = load_a_frag(arow, hi, koff);
#pragma unroll
    for (int st = 0; st < 4; ++st) {
      v16h b = load_b_frag(sfh, st * 16 + ln, hi, koff);
      acc[st] = __builtin_amdgcn_wmma_f32_16x16x32_f16(
          false, a, false, b, (short)0, acc[st], false, false);
    }
  }
#pragma unroll
  for (int st = 0; st < 4; ++st)
#pragma unroll
    for (int r = 0; r < 8; ++r) {
      int m = mtile * 16 + r + hi * 8;
      int s = st * 16 + ln;
      D[m * S_DIM + s] = acc[st][r];
    }
}

// ---------------- K6: big WMMA GEMM + fused masked exp-sum ------------------
// scores[p,s] = <vf_b[p], sf[s]>; accumulate exp(score/T) per s, zeroing
// entries where s belongs to video b and iou2d[s,p] > NEG_IOU (positives).
__global__ __launch_bounds__(256) void gemm_query(const _Float16* __restrict__ vfh,
                                                  const _Float16* __restrict__ sfh,
                                                  const float* __restrict__ iou2d,
                                                  const int* __restrict__ flat_rc,
                                                  const int* __restrict__ b2s,
                                                  const int* __restrict__ scnt,
                                                  float* __restrict__ partial) {
  __shared__ _Float16 sfs[S_DIM * SF_LDS_STRIDE];  // 33 KB staged B matrix
  __shared__ float wacc[8][S_DIM];
  int wave = threadIdx.x >> 5;
  int lane = threadIdx.x & 31;
  int hi   = lane >> 4;
  int ln   = lane & 15;
  int b    = blockIdx.x / GRP;
  int grp  = blockIdx.x % GRP;
  int ptile = grp * 8 + wave;
  int sstart = b2s[b];
  int send   = sstart + scnt[b];

  stage_sf_lds(sfh, sfs);                  // TDM async copy + tensorcnt wait

  if (ptile < PT) {                        // wave-uniform: EXEC stays all-ones
    const _Float16* arow =
        vfh + ((size_t)b * P_DIM + ptile * 16 + ln) * C_DIM;
    v8f acc[4] = {};
    for (int kk = 0; kk < C_DIM / 32; ++kk) {
      int koff = kk * 32;
      v16h a = load_a_frag(arow, hi, koff);
#pragma unroll
      for (int st = 0; st < 4; ++st) {
        v16h bm = load_b_lds(sfs, st * 16 + ln, hi, koff);
        acc[st] = __builtin_amdgcn_wmma_f32_16x16x32_f16(
            false, a, false, bm, (short)0, acc[st], false, false);
      }
    }
    // epilogue: exp(/T), mask own-video positives, reduce 16 rows per column
#pragma unroll
    for (int st = 0; st < 4; ++st) {
      int s = st * 16 + ln;
      bool own = (s >= sstart) && (s < send);
      float sum = 0.f;
#pragma unroll
      for (int r = 0; r < 8; ++r) {
        int p = ptile * 16 + r + hi * 8;
        float v = __expf(acc[st][r] * INV_T);
        if (own) {
          float iou = iou2d[(size_t)s * NN + flat_rc[p]];
          if (iou > NEG_IOU) v = 0.f;      // positive -> excluded from neg sum
        }
        sum += v;
      }
      sum += __shfl_xor(sum, 16, 32);      // combine row halves (same s)
      if (hi == 0) wacc[wave][s] = sum;
    }
  } else {
    if (hi == 0)
#pragma unroll
      for (int st = 0; st < 4; ++st) wacc[wave][st * 16 + ln] = 0.f;
  }
  __syncthreads();
  int t = threadIdx.x;
  if (t < S_DIM) {
    float tot = 0.f;
#pragma unroll
    for (int w = 0; w < 8; ++w) tot += wacc[w][t];
    partial[(size_t)blockIdx.x * S_DIM + t] = tot;
  }
}

// ---------------- K7: fixed-order reduce of partials ------------------------
__global__ __launch_bounds__(64) void reduce_partial(const float* __restrict__ partial,
                                                     float* __restrict__ negq) {
  int s = threadIdx.x;
  float tot = 0.f;
  for (int g = 0; g < NBLK; ++g) tot += partial[(size_t)g * S_DIM + s];
  negq[s] = tot;
}

// ---------------- K8: final log-CE losses + mean ----------------------------
__global__ __launch_bounds__(128) void final_loss(const float* __restrict__ D,
                                                  const float* __restrict__ negq,
                                                  const int* __restrict__ m2s,
                                                  float* __restrict__ out) {
  __shared__ float red[128];
  int m = threadIdx.x;
  int s = m2s[m];
  float pos = D[m * S_DIM + s];
  float pl  = pos * INV_T;
  float pe  = __expf(pl);
  float neg1 = 0.f;
  for (int ss = 0; ss < S_DIM; ++ss)
    if (ss != s) neg1 += __expf(D[m * S_DIM + ss] * INV_T);
  float l1 = __logf(pe + neg1)    - pl;    // inter-video
  float l2 = __logf(pe + negq[s]) - pl;    // inter-query
  red[m] = l1 + l2;
  __syncthreads();
  for (int off = 64; off > 0; off >>= 1) {
    if (m < off) red[m] += red[m + off];
    __syncthreads();
  }
  if (m == 0) out[0] = red[0] * (1.0f / (float)M_DIM);
}

// ---------------- launcher ---------------------------------------------------
extern "C" void kernel_launch(void* const* d_in, const int* in_sizes, int n_in,
                              void* d_out, int out_size, void* d_ws, size_t ws_size,
                              hipStream_t stream) {
  const float* video_feats = (const float*)d_in[0];
  const float* sents_feats = (const float*)d_in[1];
  const float* iou2d       = (const float*)d_in[2];
  const float* iou2ds      = (const float*)d_in[3];
  const int*   num_sent    = (const int*)d_in[4];
  const int*   num_tgt     = (const int*)d_in[5];
  // mask2d (d_in[6]) is statically triu(64,64); reconstructed on device.

  char* ws = (char*)d_ws;
  int*       flat_rc = (int*)(ws + OFF_FLATRC);
  int*       b2s     = (int*)(ws + OFF_B2S);
  int*       scnt    = (int*)(ws + OFF_SCNT);
  int*       m2s     = (int*)(ws + OFF_M2S);
  int*       topk    = (int*)(ws + OFF_TOPK);
  float*     Dmat    = (float*)(ws + OFF_DMAT);
  float*     partial = (float*)(ws + OFF_PARTIAL);
  float*     negq    = (float*)(ws + OFF_NEGQ);
  _Float16*  sfh     = (_Float16*)(ws + OFF_SFH);
  _Float16*  tvh     = (_Float16*)(ws + OFF_TVH);
  _Float16*  vfh     = (_Float16*)(ws + OFF_VFH);
  float*     out     = (float*)d_out;

  init_meta<<<1, 1, 0, stream>>>(num_sent, num_tgt, flat_rc, b2s, scnt, m2s);
  norm_sents<<<S_DIM, 256, 0, stream>>>(sents_feats, sfh);
  norm_video<<<B_DIM * P_DIM, 256, 0, stream>>>(video_feats, b2s, flat_rc, vfh);
  argmax_iou<<<M_DIM, 256, 0, stream>>>(iou2ds, flat_rc, topk);
  norm_topk<<<M_DIM, 256, 0, stream>>>(video_feats, m2s, topk, tvh);
  gemm_topk<<<1, 256, 0, stream>>>(tvh, sfh, Dmat);
  gemm_query<<<NBLK, 256, 0, stream>>>(vfh, sfh, iou2d, flat_rc, b2s, scnt, partial);
  reduce_partial<<<1, 64, 0, stream>>>(partial, negq);
  final_loss<<<1, 128, 0, stream>>>(Dmat, negq, m2s, out);
}
